// PosTransformer_83932250898494
// MI455X (gfx1250) — matrix operand order repaired and checked by reading, not compiled, not run on hardware
//
#include <hip/hip_runtime.h>
#include <hip/hip_bf16.h>
#include <math.h>

#define NB   16
#define NPTS 2048
#define DCH  384
#define KPAD 64      // QKV inner dim 60 padded to 64 for WMMA

typedef __attribute__((ext_vector_type(16))) _Float16 v16h;
typedef __attribute__((ext_vector_type(8)))  _Float16 v8h;
typedef __attribute__((ext_vector_type(8)))  float    v8f;

__device__ __forceinline__ v8f wmma_f16(v16h a, v16h b, v8f c) {
    // D = A(16x32 f16) * B(32x16 f16) + C(16x16 f32)
    return __builtin_amdgcn_wmma_f32_16x16x32_f16(
        /*neg_a=*/false, a, /*neg_b=*/false, b,
        /*c_mod=*/(short)0, c, /*reuse_a=*/false, /*reuse_b=*/false);
}

// A-operand (16x32 f16) register layout load: half h -> k = hi*16 + h (16 contiguous).
// rowbase points at element k=0 of this lane's row; works for global or LDS.
__device__ __forceinline__ v16h load_a(const _Float16* rowbase, int kbase, int hi) {
    union { v16h v; v8h g[2]; } u;
    u.g[0] = *(const v8h*)(rowbase + kbase + hi * 16);
    u.g[1] = *(const v8h*)(rowbase + kbase + hi * 16 + 8);
    return u.v;
}

// B-operand (32x16 f16) layout load: half h -> k = hi*16 + h (16 contiguous).
__device__ __forceinline__ v16h load_b(const _Float16* colbase, int kbase, int hi) {
    return *(const v16h*)(colbase + kbase + hi * 16);
}

// ---------------------------------------------------------------------------
// Kernel 0: convert weights to f16 (QKV weights K-padded 60 -> 64).
// ---------------------------------------------------------------------------
__global__ __launch_bounds__(256) void cvt_kernel(
    const float* __restrict__ wq, const float* __restrict__ wk,
    const float* __restrict__ wv, const float* __restrict__ w2,
    _Float16* __restrict__ wqh, _Float16* __restrict__ wkh,
    _Float16* __restrict__ wvh, _Float16* __restrict__ w2h)
{
    int idx = blockIdx.x * 256 + threadIdx.x;
    if (idx < DCH * KPAD) {
        int row = idx >> 6, c = idx & 63;
        float q = 0.f, k = 0.f, v = 0.f;
        if (c < 60) {
            q = wq[row * 60 + c];
            k = wk[row * 60 + c];
            v = wv[row * 60 + c];
        }
        wqh[idx] = (_Float16)q;
        wkh[idx] = (_Float16)k;
        wvh[idx] = (_Float16)v;
    }
    if (idx < DCH * DCH) w2h[idx] = (_Float16)w2[idx];
}

// ---------------------------------------------------------------------------
// Kernel 1: positional encoding + QKV projections, all-WMMA.
// WG = 6 waves, 32 points; wave w owns d-slice [64w, 64w+64).
// Q/K: M=points, N=d  -> coalesced (N,D) f16 stores.
// V  : M=d, N=points  -> coalesced (D,N) f16 stores.
// ---------------------------------------------------------------------------
__global__ __launch_bounds__(192) void qkv_kernel(
    const float* __restrict__ pos,
    const _Float16* __restrict__ wqh, const float* __restrict__ bq,
    const _Float16* __restrict__ wkh, const float* __restrict__ bkp,
    const _Float16* __restrict__ wvh, const float* __restrict__ bvp,
    _Float16* __restrict__ qt, _Float16* __restrict__ kt,
    _Float16* __restrict__ vm)
{
    __shared__ _Float16 semb[32][KPAD];      // f16 embedding tile, 4 KB
    const int tid  = threadIdx.x;
    const int lane = tid & 31;
    const int wave = tid >> 5;
    const int b    = blockIdx.x >> 6;        // 64 blocks of 32 points per batch
    const int n0   = (blockIdx.x & 63) << 5;
    const int mrow = lane & 15;
    const int hi   = lane >> 4;
    const int d0   = wave * 64;

    // Phase 1: NeRF encoding (channels 60..63 zero-padded)
    for (int idx = tid; idx < 32 * KPAD; idx += 192) {
        int p = idx >> 6, ch = idx & 63;
        float val = 0.f;
        if (ch < 60) {
            int coord = ch / 20, r = ch % 20;
            float x   = pos[((size_t)b * 3 + coord) * NPTS + n0 + p];
            float ang = x * (float)(1 << (r % 10));
            val = (r < 10) ? __sinf(ang) : __cosf(ang);
        }
        semb[p][ch] = (_Float16)val;
    }
    __syncthreads();

    // Hoisted embedding A-operands (M=points): [kstep][mtile]
    v16h ae[2][2];
    #pragma unroll
    for (int ks = 0; ks < 2; ++ks)
        #pragma unroll
        for (int mt = 0; mt < 2; ++mt)
            ae[ks][mt] = load_a(&semb[mt * 16 + mrow][0], ks * 32, hi);

    const float qscale = 0.051031036307982884f; // 384^-0.5 (folded softmax scale)

    // ---- Q ----
    {
        v8f C[2][4] = {};
        #pragma unroll
        for (int ks = 0; ks < 2; ++ks) {
            v16h bw[4];
            #pragma unroll
            for (int nt = 0; nt < 4; ++nt)
                bw[nt] = load_b(wqh + (size_t)(d0 + nt * 16 + mrow) * KPAD, ks * 32, hi);
            #pragma unroll
            for (int mt = 0; mt < 2; ++mt)
                #pragma unroll
                for (int nt = 0; nt < 4; ++nt)
                    C[mt][nt] = wmma_f16(ae[ks][mt], bw[nt], C[mt][nt]);
        }
        #pragma unroll
        for (int nt = 0; nt < 4; ++nt) {
            float bias = bq[d0 + nt * 16 + mrow];
            #pragma unroll
            for (int mt = 0; mt < 2; ++mt)
                #pragma unroll
                for (int r = 0; r < 8; ++r) {
                    int p = n0 + mt * 16 + r + hi * 8;
                    qt[((size_t)(b * NPTS + p)) * DCH + d0 + nt * 16 + mrow] =
                        (_Float16)((C[mt][nt][r] + bias) * qscale);
                }
        }
    }
    // ---- K ----
    {
        v8f C[2][4] = {};
        #pragma unroll
        for (int ks = 0; ks < 2; ++ks) {
            v16h bw[4];
            #pragma unroll
            for (int nt = 0; nt < 4; ++nt)
                bw[nt] = load_b(wkh + (size_t)(d0 + nt * 16 + mrow) * KPAD, ks * 32, hi);
            #pragma unroll
            for (int mt = 0; mt < 2; ++mt)
                #pragma unroll
                for (int nt = 0; nt < 4; ++nt)
                    C[mt][nt] = wmma_f16(ae[ks][mt], bw[nt], C[mt][nt]);
        }
        #pragma unroll
        for (int nt = 0; nt < 4; ++nt) {
            float bias = bkp[d0 + nt * 16 + mrow];
            #pragma unroll
            for (int mt = 0; mt < 2; ++mt)
                #pragma unroll
                for (int r = 0; r < 8; ++r) {
                    int p = n0 + mt * 16 + r + hi * 8;
                    kt[((size_t)(b * NPTS + p)) * DCH + d0 + nt * 16 + mrow] =
                        (_Float16)(C[mt][nt][r] + bias);
                }
        }
    }
    // ---- V (M=d, N=points) ----
    {
        v8f C[4][2] = {};
        #pragma unroll
        for (int ks = 0; ks < 2; ++ks) {
            v16h aw[4];
            #pragma unroll
            for (int mt = 0; mt < 4; ++mt)
                aw[mt] = load_a(wvh + (size_t)(d0 + mt * 16 + mrow) * KPAD, ks * 32, hi);
            v16h be[2];
            #pragma unroll
            for (int pt = 0; pt < 2; ++pt)
                be[pt] = load_b(&semb[pt * 16 + mrow][0], ks * 32, hi);
            #pragma unroll
            for (int mt = 0; mt < 4; ++mt)
                #pragma unroll
                for (int pt = 0; pt < 2; ++pt)
                    C[mt][pt] = wmma_f16(aw[mt], be[pt], C[mt][pt]);
        }
        #pragma unroll
        for (int mt = 0; mt < 4; ++mt)
            #pragma unroll
            for (int r = 0; r < 8; ++r) {
                int d = d0 + mt * 16 + r + hi * 8;
                float bias = bvp[d];
                #pragma unroll
                for (int pt = 0; pt < 2; ++pt)
                    vm[((size_t)b * DCH + d) * NPTS + n0 + pt * 16 + mrow] =
                        (_Float16)(C[mt][pt][r] + bias);
            }
    }
}

// ---------------------------------------------------------------------------
// Kernel 2: residual conv1x1(3->384) -> BN -> ReLU -> WMMA conv1x1(384->384).
// WG = 6 waves, 32 points; wave w owns output d-slice [64w, 64w+64).
// ---------------------------------------------------------------------------
__global__ __launch_bounds__(192) void res_kernel(
    const float* __restrict__ pos,
    const float* __restrict__ w1, const float* __restrict__ b1,
    const float* __restrict__ gamma, const float* __restrict__ beta,
    const float* __restrict__ mean,  const float* __restrict__ var,
    const _Float16* __restrict__ w2h, const float* __restrict__ b2,
    float* __restrict__ out)
{
    __shared__ _Float16 sh[32][DCH];         // f16 hidden tile, 24 KB
    const int tid  = threadIdx.x;
    const int lane = tid & 31;
    const int wave = tid >> 5;
    const int b    = blockIdx.x >> 6;        // 64 blocks of 32 points per batch
    const int n0   = (blockIdx.x & 63) << 5;
    const int mrow = lane & 15;
    const int hi   = lane >> 4;
    const int d0   = wave * 64;

    // Phase 1: hidden = relu(BN(w1 @ pos + b1)) -> f16 LDS (stored (p, c))
    for (int d = tid; d < DCH; d += 192) {
        float w10 = w1[d * 3 + 0], w11 = w1[d * 3 + 1], w12 = w1[d * 3 + 2];
        float inv = gamma[d] * rsqrtf(var[d] + 1e-5f);
        float mu  = mean[d], bt = beta[d], b1d = b1[d];
        for (int p = 0; p < 32; ++p) {
            float x0 = pos[((size_t)b * 3 + 0) * NPTS + n0 + p];
            float x1 = pos[((size_t)b * 3 + 1) * NPTS + n0 + p];
            float x2 = pos[((size_t)b * 3 + 2) * NPTS + n0 + p];
            float r1 = w10 * x0 + w11 * x1 + w12 * x2 + b1d;
            float h  = (r1 - mu) * inv + bt;
            sh[p][d] = (_Float16)fmaxf(h, 0.f);
        }
    }
    __syncthreads();

    // Phase 2: out[d, p] = w2[d,:] @ hidden[:, p] + b2[d]   (M=d, N=points)
    v8f C[4][2] = {};
    for (int c0 = 0; c0 < DCH; c0 += 32) {
        v16h aw[4];
        #pragma unroll
        for (int mt = 0; mt < 4; ++mt)
            aw[mt] = load_a(w2h + (size_t)(d0 + mt * 16 + mrow) * DCH, c0, hi);
        v16h bh[2];
        #pragma unroll
        for (int pt = 0; pt < 2; ++pt)
            bh[pt] = load_b(&sh[pt * 16 + mrow][0], c0, hi);
        #pragma unroll
        for (int mt = 0; mt < 4; ++mt)
            #pragma unroll
            for (int pt = 0; pt < 2; ++pt)
                C[mt][pt] = wmma_f16(aw[mt], bh[pt], C[mt][pt]);
    }
    #pragma unroll
    for (int mt = 0; mt < 4; ++mt)
        #pragma unroll
        for (int r = 0; r < 8; ++r) {
            int d = d0 + mt * 16 + r + hi * 8;
            float bias = b2[d];
            #pragma unroll
            for (int pt = 0; pt < 2; ++pt)
                out[((size_t)b * DCH + d) * NPTS + n0 + pt * 16 + mrow] =
                    C[mt][pt][r] + bias;
        }
}

// ---------------------------------------------------------------------------
// Kernel 3: flash attention, WMMA f32_16x16x32_f16.
// WG = 6 waves, 16 query rows; wave w owns D-slice [64w, 64w+64).
// Per 32-col step: partial QK^T (4 wmma) -> ds_add_f32 reduce into ping-pong
// LDS tile -> online softmax (registers + shfl) -> P packed in A-layout ->
// P*V (4 wmma) into f32 accumulators. 2 barriers / iteration.
// ---------------------------------------------------------------------------
__global__ __launch_bounds__(192) void attn_kernel(
    const _Float16* __restrict__ qt,   // (B, N, D) f16, pre-scaled
    const _Float16* __restrict__ kt,   // (B, N, D) f16
    const _Float16* __restrict__ vm,   // (B, D, N) f16
    float* __restrict__ out)           // (B, D, N) f32, pre-filled with residual
{
    __shared__ float sS[2 * 512];      // two 16x32 f32 score tiles (ping-pong)

    const int tid  = threadIdx.x;
    const int lane = tid & 31;
    const int wave = tid >> 5;         // 0..5
    const int b    = blockIdx.x >> 7;  // N/16 = 128 query tiles per batch
    const int i0   = (blockIdx.x & 127) << 4;
    const int d0   = wave * 64;
    const int mrow = lane & 15;
    const int hi   = lane >> 4;

    for (int idx = tid; idx < 1024; idx += 192) sS[idx] = 0.f;

    // Hoisted Q A-operands for this wave's D slice (rows i0..i0+15).
    v16h aq0, aq1;
    {
        const _Float16* qrow = qt + ((size_t)(b * NPTS + i0 + mrow)) * DCH;
        aq0 = load_a(qrow, d0, hi);
        aq1 = load_a(qrow, d0 + 32, hi);
    }

    v8f o0 = {}, o1 = {}, o2 = {}, o3 = {};
    float m_run = -INFINITY, l_run = 0.f;
    __syncthreads();

    for (int j0 = 0, it = 0; j0 < NPTS; j0 += 32, ++it) {
        float* Sb = &sS[(it & 1) * 512];

        // partial S = Q_slice^T * K_slice : two 16x16 C tiles
        const _Float16* kb = kt + ((size_t)(b * NPTS + j0)) * DCH + d0 + hi * 16;
        v8f sc0 = {}, sc1 = {};
        {
            v16h bk0 = *(const v16h*)(kb + (size_t)mrow * DCH);
            sc0 = wmma_f16(aq0, bk0, sc0);
            v16h bk1 = *(const v16h*)(kb + (size_t)mrow * DCH + 32);
            sc0 = wmma_f16(aq1, bk1, sc0);
            v16h bk2 = *(const v16h*)(kb + (size_t)(16 + mrow) * DCH);
            sc1 = wmma_f16(aq0, bk2, sc1);
            v16h bk3 = *(const v16h*)(kb + (size_t)(16 + mrow) * DCH + 32);
            sc1 = wmma_f16(aq1, bk3, sc1);
        }
        #pragma unroll
        for (int r = 0; r < 8; ++r) {
            atomicAdd(&Sb[(r + hi * 8) * 32 + mrow],      sc0[r]);
            atomicAdd(&Sb[(r + hi * 8) * 32 + 16 + mrow], sc1[r]);
        }
        __syncthreads();

        // read full S row (A-layout half mapping) + zero other buffer
        float s[16];
        #pragma unroll
        for (int h = 0; h < 16; ++h) {
            int k = (hi << 4) + h;
            s[h] = Sb[mrow * 32 + k];
        }
        {
            float* Zb = &sS[((it & 1) ^ 1) * 512];
            for (int idx = tid; idx < 512; idx += 192) Zb[idx] = 0.f;
        }

        // online softmax (redundant per wave; identical results)
        float tmax = s[0];
        #pragma unroll
        for (int h = 1; h < 16; ++h) tmax = fmaxf(tmax, s[h]);
        tmax = fmaxf(tmax, __shfl_xor(tmax, 16, 32));
        float m_new = fmaxf(m_run, tmax);
        float p[16], tsum = 0.f;
        #pragma unroll
        for (int h = 0; h < 16; ++h) { p[h] = __expf(s[h] - m_new); tsum += p[h]; }
        tsum += __shfl_xor(tsum, 16, 32);
        float alpha = __expf(m_run - m_new);
        l_run = l_run * alpha + tsum;
        m_run = m_new;

        #pragma unroll
        for (int r = 0; r < 8; ++r) {
            float ar = __shfl(alpha, r + hi * 8, 32);
            o0[r] *= ar; o1[r] *= ar; o2[r] *= ar; o3[r] *= ar;
        }

        union { v16h v; _Float16 h16[16]; } up;
        #pragma unroll
        for (int h = 0; h < 16; ++h) up.h16[h] = (_Float16)p[h];

        // O += P * V_slice : 4 wmma
        const _Float16* vb = vm + ((size_t)b * DCH + d0 + mrow) * NPTS + j0 + hi * 16;
        o0 = wmma_f16(up.v, *(const v16h*)(vb + 0 * 16 * NPTS), o0);
        o1 = wmma_f16(up.v, *(const v16h*)(vb + 1 * 16 * NPTS), o1);
        o2 = wmma_f16(up.v, *(const v16h*)(vb + 2 * 16 * NPTS), o2);
        o3 = wmma_f16(up.v, *(const v16h*)(vb + 3 * 16 * NPTS), o3);
        __syncthreads();
    }

    // epilogue: divide by row sums, add into residual in d_out
    float* ob = out + (size_t)b * DCH * NPTS;
    #pragma unroll
    for (int r = 0; r < 8; ++r) {
        float lr  = __shfl(l_run, r + hi * 8, 32);
        float inv = 1.f / lr;
        int   irow = i0 + r + hi * 8;
        ob[(size_t)(d0 +  0 + mrow) * NPTS + irow] += o0[r] * inv;
        ob[(size_t)(d0 + 16 + mrow) * NPTS + irow] += o1[r] * inv;
        ob[(size_t)(d0 + 32 + mrow) * NPTS + irow] += o2[r] * inv;
        ob[(size_t)(d0 + 48 + mrow) * NPTS + irow] += o3[r] * inv;
    }
}

// ---------------------------------------------------------------------------
extern "C" void kernel_launch(void* const* d_in, const int* in_sizes, int n_in,
                              void* d_out, int out_size, void* d_ws, size_t ws_size,
                              hipStream_t stream) {
    const float* pos   = (const float*)d_in[0];
    const float* wq    = (const float*)d_in[1];
    const float* bq    = (const float*)d_in[2];
    const float* wk    = (const float*)d_in[3];
    const float* bk    = (const float*)d_in[4];
    const float* wv    = (const float*)d_in[5];
    const float* bv    = (const float*)d_in[6];
    const float* w1    = (const float*)d_in[7];
    const float* b1    = (const float*)d_in[8];
    const float* gamma = (const float*)d_in[9];
    const float* beta  = (const float*)d_in[10];
    const float* mean  = (const float*)d_in[11];
    const float* var   = (const float*)d_in[12];
    const float* w2    = (const float*)d_in[13];
    const float* b2    = (const float*)d_in[14];
    float* out = (float*)d_out;

    const size_t elems = (size_t)NB * NPTS * DCH;   // 12.58M halfs
    _Float16* qt  = (_Float16*)d_ws;
    _Float16* kt  = qt  + elems;
    _Float16* vm  = kt  + elems;
    _Float16* wqh = vm  + elems;                     // 384*64
    _Float16* wkh = wqh + (size_t)DCH * KPAD;
    _Float16* wvh = wkh + (size_t)DCH * KPAD;
    _Float16* w2h = wvh + (size_t)DCH * KPAD;        // 384*384

    cvt_kernel<<<(DCH * DCH + 255) / 256, 256, 0, stream>>>(
        wq, wk, wv, w2, wqh, wkh, wvh, w2h);
    qkv_kernel<<<NB * (NPTS / 32), 192, 0, stream>>>(
        pos, wqh, bq, wkh, bk, wvh, bv, qt, kt, vm);
    res_kernel<<<NB * (NPTS / 32), 192, 0, stream>>>(
        pos, w1, b1, gamma, beta, mean, var, w2h, b2, out);
    attn_kernel<<<NB * (NPTS / 16), 192, 0, stream>>>(qt, kt, vm, out);
}
